// EGNNLayer_36335423324795
// MI455X (gfx1250) — compile-verified, matrix operand order checked by reading, not run on hardware
//
#include <hip/hip_runtime.h>

typedef __attribute__((ext_vector_type(16))) __bf16 v16bf;
typedef __attribute__((ext_vector_type(8)))  float  v8f;

union Frag { v16bf v; uint4 q[2]; };

#define H  128
#define SQ 17           // padded row stride in uint4 (16 data + 1 pad) -> no LDS bank conflicts
#define SE (SQ * 8)     // padded row stride in bf16 elements
#define ET 64           // edges (or nodes) per block = 4 subtiles of 16

#if defined(__gfx1250__)
#define ASYNC_GATHER 1
#else
#define ASYNC_GATHER 0
#endif

__device__ __forceinline__ float fast_silu(float v) {
    return v * (1.0f / (1.0f + __expf(-v)));
}

// 16B gather: global -> LDS. On gfx1250 use the async-to-LDS path (ASYNCcnt,
// no VGPR staging); otherwise plain copy.
__device__ __forceinline__ void gather16(const uint4* g, uint4* l) {
#if ASYNC_GATHER
    unsigned ldsoff = (unsigned)(size_t)l;   // LDS addr = generic addr[31:0]
    asm volatile("global_load_async_to_lds_b128 %0, %1, off"
                 :: "v"(ldsoff), "v"(g) : "memory");
#else
    *l = *g;
#endif
}

__device__ __forceinline__ void gather_fence() {
#if ASYNC_GATHER
    asm volatile("s_wait_asynccnt 0" ::: "memory");
#endif
}

// ---------------- prep kernels ----------------

__global__ void cvt_x_kernel(const float* __restrict__ x, __bf16* __restrict__ xb, long long n) {
    long long i = (long long)blockIdx.x * 256 + threadIdx.x;
    if (i < n) xb[i] = (__bf16)x[i];
}

__global__ void zero_kernel(float* __restrict__ p, long long n) {
    long long i = (long long)blockIdx.x * 256 + threadIdx.x;
    if (i < n) p[i] = 0.0f;
}

// dst[n][k] = bf16(src[k][n]);  src is [K][128] row-major (only first K rows used)
__global__ void cvt_transpose_kernel(const float* __restrict__ src, __bf16* __restrict__ dst, int K) {
    int idx = blockIdx.x * 256 + threadIdx.x;
    if (idx < 128 * K) {
        int n = idx / K;
        int k = idx - n * K;
        dst[(size_t)n * K + k] = (__bf16)src[(size_t)k * H + n];
    }
}

// ---------------- edge kernel: message MLP + scatter-add ----------------
// 64 edges per block (4 subtiles of 16), 256 threads = 8 waves.
// Wave w owns output columns [16w,16w+16); B-fragments are loaded once per
// K-step and reused across the 4 edge subtiles (4 accumulators).

__global__ __launch_bounds__(256) void egnn_edge_kernel(
    const __bf16* __restrict__ xbf, const float* __restrict__ pos,
    const int* __restrict__ eidx, long long E,
    const __bf16* __restrict__ w1t,       // [128][256] bf16, k-contiguous
    const float*  __restrict__ w1raw,     // msg_w1 raw [257][128] f32 (row 256 = dist weights)
    const float*  __restrict__ b1,
    const __bf16* __restrict__ w2t,       // [128][128] bf16
    const float*  __restrict__ b2,
    float* __restrict__ aggr)
{
    __shared__ uint4 sXs[ET * SQ];
    __shared__ uint4 sXr[ET * SQ];
    __shared__ uint4 sH [ET * SQ];
    __shared__ int   sSend[ET];
    __shared__ int   sRec [ET];
    __shared__ float sDist[ET];

    const int t = threadIdx.x;
    const long long ebase = (long long)blockIdx.x * ET;

    if (t < ET) {
        long long e = ebase + t;
        int s = 0, r = -1;
        float d = 0.0f;
        if (e < E) {
            s = eidx[e];
            r = eidx[E + e];
            float dx = pos[3 * s + 0] - pos[3 * r + 0];
            float dy = pos[3 * s + 1] - pos[3 * r + 1];
            float dz = pos[3 * s + 2] - pos[3 * r + 2];
            d = sqrtf(dx * dx + dy * dy + dz * dz);
        }
        sSend[t] = s; sRec[t] = r; sDist[t] = d;
    }
    __syncthreads();

    {   // gather x[send], x[rec] rows (bf16, 256B each) into LDS, async path
        const int r = t >> 4, c = t & 15;
        const uint4* xq = (const uint4*)xbf;          // row stride = 16 uint4
        #pragma unroll
        for (int st = 0; st < 4; ++st) {
            int er = st * 16 + r;
            int si = sSend[er];
            int ri = sRec[er] < 0 ? 0 : sRec[er];
            gather16(xq + (size_t)si * 16 + c, &sXs[er * SQ + c]);
            gather16(xq + (size_t)ri * 16 + c, &sXr[er * SQ + c]);
        }
    }
    gather_fence();
    __syncthreads();

    const int lane  = t & 31;
    const int wave  = t >> 5;
    const int nbase = wave * 16;
    const int m     = lane & 15;
    const int hi16  = lane >> 4;           // 0 or 1
    const int kb    = hi16 ? 8 : 0;        // per-lane K base (bf16 elems)
    const int ncol  = nbase + m;

    // ---- GEMM1: [64x256] x [256x16] ; B loaded once per K-step ----
    v8f acc[4] = {{}, {}, {}, {}};
    #pragma unroll
    for (int kk = 0; kk < 8; ++kk) {
        Frag b;
        const uint4* wq = (const uint4*)(w1t + (size_t)ncol * 256);
        const int kcol = kk * 32;
        b.q[0] = wq[(kcol + kb) >> 3];
        b.q[1] = wq[(kcol + 16 + kb) >> 3];
        const uint4* tq = (kk < 4) ? sXs : sXr;
        const int krow = (kk & 3) * 32;
        #pragma unroll
        for (int st = 0; st < 4; ++st) {
            Frag a;
            a.q[0] = tq[(st * 16 + m) * SQ + ((krow + kb) >> 3)];
            a.q[1] = tq[(st * 16 + m) * SQ + ((krow + 16 + kb) >> 3)];
            acc[st] = __builtin_amdgcn_wmma_f32_16x16x32_bf16(false, a.v, false, b.v,
                                                              (short)0, acc[st], false, false);
        }
    }

    // epilogue 1: + dist * w1[256,:] + bias, SiLU -> bf16 LDS tile
    {
        const float wd = w1raw[256 * H + ncol];
        const float bb = b1[ncol];
        __bf16* hT = (__bf16*)sH;
        #pragma unroll
        for (int st = 0; st < 4; ++st) {
            #pragma unroll
            for (int i = 0; i < 8; ++i) {
                int row = st * 16 + i + (hi16 << 3);
                float pre = acc[st][i] + sDist[row] * wd + bb;
                hT[row * SE + ncol] = (__bf16)fast_silu(pre);
            }
        }
    }
    __syncthreads();

    // ---- GEMM2: [64x128] x [128x16] ----
    v8f acc2[4] = {{}, {}, {}, {}};
    #pragma unroll
    for (int kk = 0; kk < 4; ++kk) {
        Frag b;
        const uint4* wq = (const uint4*)(w2t + (size_t)ncol * 128);
        const int krow = kk * 32;
        b.q[0] = wq[(krow + kb) >> 3];
        b.q[1] = wq[(krow + 16 + kb) >> 3];
        #pragma unroll
        for (int st = 0; st < 4; ++st) {
            Frag a;
            a.q[0] = sH[(st * 16 + m) * SQ + ((krow + kb) >> 3)];
            a.q[1] = sH[(st * 16 + m) * SQ + ((krow + 16 + kb) >> 3)];
            acc2[st] = __builtin_amdgcn_wmma_f32_16x16x32_bf16(false, a.v, false, b.v,
                                                               (short)0, acc2[st], false, false);
        }
    }

    // epilogue 2: bias + SiLU, scatter-add into aggr (L2-resident) via f32 atomics
    {
        const float bb = b2[ncol];
        #pragma unroll
        for (int st = 0; st < 4; ++st) {
            #pragma unroll
            for (int i = 0; i < 8; ++i) {
                int row = st * 16 + i + (hi16 << 3);
                int rec = sRec[row];
                if (rec >= 0) {
                    float msg = fast_silu(acc2[st][i] + bb);
                    float* p = aggr + (size_t)rec * H + ncol;
                    asm volatile("global_atomic_add_f32 %0, %1, off"
                                 :: "v"(p), "v"(msg) : "memory");
                }
            }
        }
    }
}

// ---------------- node kernel: update MLP ----------------
// 64 nodes per block, same register-blocked structure.

__global__ __launch_bounds__(256) void egnn_node_kernel(
    const __bf16* __restrict__ xbf, const float* __restrict__ aggr,
    const __bf16* __restrict__ u1t,       // [128][256] bf16
    const float*  __restrict__ ub1,
    const __bf16* __restrict__ u2t,       // [128][128] bf16
    const float*  __restrict__ ub2,
    float* __restrict__ out, int N)
{
    __shared__ uint4 sX[ET * SQ];
    __shared__ uint4 sA[ET * SQ];
    __shared__ uint4 sH[ET * SQ];

    const int t = threadIdx.x;
    const int base = blockIdx.x * ET;

    {   // x tile (bf16) via async gather; aggr tile (f32->bf16) converted in VALU
        const int r = t >> 4, c = t & 15;
        const uint4* xq = (const uint4*)xbf;
        #pragma unroll
        for (int st = 0; st < 4; ++st) {
            int row = st * 16 + r;
            int node = base + row; if (node >= N) node = N - 1;
            gather16(xq + (size_t)node * 16 + c, &sX[row * SQ + c]);
        }
        __bf16* aT = (__bf16*)sA;
        #pragma unroll
        for (int j = 0; j < 32; ++j) {
            int idx = j * 256 + t;           // 0..8191, coalesced along columns
            int row = idx >> 7, col = idx & 127;
            int node = base + row; if (node >= N) node = N - 1;
            aT[row * SE + col] = (__bf16)aggr[(size_t)node * H + col];
        }
    }
    gather_fence();
    __syncthreads();

    const int lane  = t & 31;
    const int wave  = t >> 5;
    const int nbase = wave * 16;
    const int m     = lane & 15;
    const int hi16  = lane >> 4;
    const int kb    = hi16 ? 8 : 0;
    const int ncol  = nbase + m;

    // GEMM1: concat(x, aggr)[64x256] x u1t
    v8f acc[4] = {{}, {}, {}, {}};
    #pragma unroll
    for (int kk = 0; kk < 8; ++kk) {
        Frag b;
        const uint4* wq = (const uint4*)(u1t + (size_t)ncol * 256);
        const int kcol = kk * 32;
        b.q[0] = wq[(kcol + kb) >> 3];
        b.q[1] = wq[(kcol + 16 + kb) >> 3];
        const uint4* tq = (kk < 4) ? sX : sA;
        const int krow = (kk & 3) * 32;
        #pragma unroll
        for (int st = 0; st < 4; ++st) {
            Frag a;
            a.q[0] = tq[(st * 16 + m) * SQ + ((krow + kb) >> 3)];
            a.q[1] = tq[(st * 16 + m) * SQ + ((krow + 16 + kb) >> 3)];
            acc[st] = __builtin_amdgcn_wmma_f32_16x16x32_bf16(false, a.v, false, b.v,
                                                              (short)0, acc[st], false, false);
        }
    }

    {
        const float bb = ub1[ncol];
        __bf16* hT = (__bf16*)sH;
        #pragma unroll
        for (int st = 0; st < 4; ++st) {
            #pragma unroll
            for (int i = 0; i < 8; ++i) {
                int row = st * 16 + i + (hi16 << 3);
                hT[row * SE + ncol] = (__bf16)fast_silu(acc[st][i] + bb);
            }
        }
    }
    __syncthreads();

    // GEMM2
    v8f acc2[4] = {{}, {}, {}, {}};
    #pragma unroll
    for (int kk = 0; kk < 4; ++kk) {
        Frag b;
        const uint4* wq = (const uint4*)(u2t + (size_t)ncol * 128);
        const int krow = kk * 32;
        b.q[0] = wq[(krow + kb) >> 3];
        b.q[1] = wq[(krow + 16 + kb) >> 3];
        #pragma unroll
        for (int st = 0; st < 4; ++st) {
            Frag a;
            a.q[0] = sH[(st * 16 + m) * SQ + ((krow + kb) >> 3)];
            a.q[1] = sH[(st * 16 + m) * SQ + ((krow + 16 + kb) >> 3)];
            acc2[st] = __builtin_amdgcn_wmma_f32_16x16x32_bf16(false, a.v, false, b.v,
                                                               (short)0, acc2[st], false, false);
        }
    }

    {
        const float bb = ub2[ncol];
        #pragma unroll
        for (int st = 0; st < 4; ++st) {
            #pragma unroll
            for (int i = 0; i < 8; ++i) {
                int row = st * 16 + i + (hi16 << 3);
                int node = base + row;
                if (node < N) out[(size_t)node * H + ncol] = acc2[st][i] + bb;
            }
        }
    }
}

// ---------------- launch ----------------

extern "C" void kernel_launch(void* const* d_in, const int* in_sizes, int n_in,
                              void* d_out, int out_size, void* d_ws, size_t ws_size,
                              hipStream_t stream) {
    const float* x      = (const float*)d_in[0];
    const float* pos    = (const float*)d_in[1];
    const int*   eidx   = (const int*)  d_in[2];
    const float* msg_w1 = (const float*)d_in[3];
    const float* msg_b1 = (const float*)d_in[4];
    const float* msg_w2 = (const float*)d_in[5];
    const float* msg_b2 = (const float*)d_in[6];
    const float* upd_w1 = (const float*)d_in[7];
    const float* upd_b1 = (const float*)d_in[8];
    const float* upd_w2 = (const float*)d_in[9];
    const float* upd_b2 = (const float*)d_in[10];
    float* out = (float*)d_out;

    const int N = in_sizes[0] / H;
    const long long E = (long long)in_sizes[2] / 2;

    char* ws = (char*)d_ws;
    size_t off = 0;
    auto take = [&](size_t bytes) { size_t cur = off; off = (off + bytes + 255) & ~(size_t)255; return cur; };
    __bf16* xbf  = (__bf16*)(ws + take((size_t)N * H * 2));
    float*  aggr = (float*) (ws + take((size_t)N * H * 4));
    __bf16* w1t  = (__bf16*)(ws + take((size_t)128 * 256 * 2));
    __bf16* w2t  = (__bf16*)(ws + take((size_t)128 * 128 * 2));
    __bf16* u1t  = (__bf16*)(ws + take((size_t)128 * 256 * 2));
    __bf16* u2t  = (__bf16*)(ws + take((size_t)128 * 128 * 2));
    (void)ws_size; (void)n_in; (void)out_size;

    long long totX = (long long)N * H;
    int bx = (int)((totX + 255) / 256);
    cvt_x_kernel<<<bx, 256, 0, stream>>>(x, xbf, totX);
    zero_kernel <<<bx, 256, 0, stream>>>(aggr, totX);
    cvt_transpose_kernel<<<(128 * 256 + 255) / 256, 256, 0, stream>>>(msg_w1, w1t, 256);
    cvt_transpose_kernel<<<(128 * 128 + 255) / 256, 256, 0, stream>>>(msg_w2, w2t, 128);
    cvt_transpose_kernel<<<(128 * 256 + 255) / 256, 256, 0, stream>>>(upd_w1, u1t, 256);
    cvt_transpose_kernel<<<(128 * 128 + 255) / 256, 256, 0, stream>>>(upd_w2, u2t, 128);

    int etiles = (int)((E + ET - 1) / ET);
    egnn_edge_kernel<<<etiles, 256, 0, stream>>>(xbf, pos, eidx, E,
                                                 w1t, msg_w1, msg_b1, w2t, msg_b2, aggr);

    int ntiles = (N + ET - 1) / ET;
    egnn_node_kernel<<<ntiles, 256, 0, stream>>>(xbf, aggr, u1t, upd_b1, u2t, upd_b2, out, N);
}
